// BSCNUTrain_62912680951981
// MI455X (gfx1250) — matrix operand order repaired and checked by reading, not compiled, not run on hardware
//
#include <hip/hip_runtime.h>

typedef unsigned short u16;
typedef unsigned int u32;
typedef unsigned long long u64;

#define N_ROWS 30000
#define DIMR   784
#define DPAD   800      // K padded to multiple of 32
#define NCAND  100
#define NCPAD  112      // 7 tiles of 16
#define MAXK   12
#define MAXPAD 27840    // max batch (27826) padded to 16

typedef __bf16 bf16_t;
typedef bf16_t v16bf __attribute__((ext_vector_type(16)));
typedef float  v8f   __attribute__((ext_vector_type(8)));

union Frag { v16bf v; uint4 u[2]; };

// ---------- small device helpers ----------
__device__ __forceinline__ u32 pcg(u32 x) {
    x = x * 747796405u + 2891336453u;
    u32 w = ((x >> ((x >> 28u) + 4u)) ^ x) * 277803737u;
    return (w >> 22u) ^ w;
}
__device__ __forceinline__ float u01(u32 h) { return (h >> 8) * (1.0f / 16777216.0f); }
__device__ __forceinline__ u16 f2bf(float f) {
    u32 u = __float_as_uint(f);
    u32 r = u + 0x7FFFu + ((u >> 16) & 1u);   // RNE
    return (u16)(r >> 16);
}
__device__ __forceinline__ float bfl(u32 u) { return __uint_as_float(u << 16); }
__device__ __forceinline__ float bfh(u32 u) { return __uint_as_float(u & 0xFFFF0000u); }
// fast sigmoid: v_exp_f32 + v_rcp_f32 (no IEEE division sequence)
__device__ __forceinline__ float sigmoidf(float x) {
    return __builtin_amdgcn_rcpf(1.0f + __expf(-x));
}

// ---------- one-time: X (f32, 30000x784) -> bf16 padded to 800 ----------
__global__ __launch_bounds__(256) void convert_x(const float* __restrict__ X,
                                                 u16* __restrict__ Xb) {
    int row = blockIdx.x;
    for (int i = threadIdx.x; i < DPAD; i += 256) {
        float v = (i < DIMR) ? X[(size_t)row * DIMR + i] : 0.0f;
        Xb[(size_t)row * DPAD + i] = f2bf(v);
    }
}

// ---------- per-neuron: random candidate weights/biases ----------
__global__ __launch_bounds__(256) void gen_wb(u16* __restrict__ Wt,
                                              float* __restrict__ Wf,
                                              float* __restrict__ bvec, int k) {
    int t = blockIdx.x * 256 + threadIdx.x;
    if (t < NCPAD * DPAD) {
        int c = t / DPAD, i = t - c * DPAD;
        float val = 0.0f;
        if (c < NCAND && i < DIMR) {
            float lam = (c < NCAND / 2) ? 1.0f : 10.0f;
            u32 h = pcg(pcg(0x5C17A11u + (u32)k * 0x9E3779B1u) ^ (u32)(c * DPAD + i));
            val = lam * (2.0f * u01(h) - 1.0f);
            Wf[(size_t)c * DIMR + i] = val;
        }
        Wt[(size_t)c * DPAD + i] = f2bf(val);
    }
    if (t < NCPAD) {
        float bval = 0.0f;
        if (t < NCAND) {
            float lam = (t < NCAND / 2) ? 1.0f : 10.0f;
            u32 h = pcg(pcg(0xB1A5EDu + (u32)k * 0x85EBCA77u) ^ (u32)t);
            bval = lam * (2.0f * u01(h) - 1.0f);
        }
        bvec[t] = bval;
    }
}

// ---------- per-neuron: permutation sample (affine bijection mod N) ----------
__global__ __launch_bounds__(256) void gen_idx(int* __restrict__ idx, int k) {
    int t = blockIdx.x * 256 + threadIdx.x;
    if (t < MAXPAD) {
        u32 off = pcg(0x1D0u + (u32)k) % N_ROWS;
        idx[t] = (int)(((u64)7919u * (u64)t + off) % (u64)N_ROWS);
    }
}

// ---------- per-neuron: ek = Y[idx] - (H[:, :k] beta)[idx] ----------
__global__ __launch_bounds__(256) void ek_kernel(const float* __restrict__ Y,
                                                 const float* __restrict__ H,
                                                 const float* __restrict__ beta,
                                                 const int* __restrict__ idx,
                                                 float* __restrict__ ek,
                                                 int k, int batch) {
    int i = blockIdx.x * 256 + threadIdx.x;
    if (i >= batch) return;
    int row = idx[i];
    float hv[MAXK];
    for (int m = 0; m < k; ++m) hv[m] = H[(size_t)m * N_ROWS + row];
    for (int j = 0; j < 10; ++j) {
        float s = Y[(size_t)row * 10 + j];
        for (int m = 0; m < k; ++m) s -= hv[m] * beta[m * 10 + j];
        ek[(size_t)i * 10 + j] = s;
    }
}

// ---------- per-neuron: zero accumulators ----------
__global__ __launch_bounds__(256) void zero_acc(float* vnum, float* vden,
                                                float* scal, float* QT, int k) {
    int t = blockIdx.x * 256 + threadIdx.x;
    if (t < 10 * NCPAD) vnum[t] = 0.0f;
    if (t < NCPAD) vden[t] = 0.0f;
    if (t < 4) scal[t] = 0.0f;
    if (t < 10) QT[k * 10 + t] = 0.0f;
}

// ---------- candidate scoring: fused WMMA GEMM + sigmoid + reductions ----------
// grid = (7 N-tiles, MBLOCKS), 256 threads = 8 waves; each wave owns 16x16 tiles.
__global__ __launch_bounds__(256) void cand_gemm(const u16* __restrict__ Xb,
                                                 const u16* __restrict__ Wt,
                                                 const float* __restrict__ bvec,
                                                 const int* __restrict__ idx,
                                                 const float* __restrict__ ek,
                                                 float* __restrict__ vnum,
                                                 float* __restrict__ vden,
                                                 int batch) {
    __shared__ u16 Bl[16 * DPAD];                 // 25.6 KB candidate strip
    const int tid = threadIdx.x;
    const int nbase = blockIdx.x * 16;

    // Stage B strip (16 candidates x 800 K, bf16) into LDS with CDNA5 async copy.
    {
        const u16* gsrc = Wt + (size_t)nbase * DPAD;
        for (int i = tid; i < (16 * DPAD) / 8; i += 256) {   // 1600 x 16B
            u32 lds = (u32)(u64)(&Bl[i * 8]);
            u64 ga  = (u64)(gsrc + i * 8);
            asm volatile("global_load_async_to_lds_b128 %0, %1, off"
                         :: "v"(lds), "v"(ga) : "memory");
        }
        asm volatile("s_wait_asynccnt 0" ::: "memory");
    }
    __syncthreads();

    const int wave = tid >> 5;
    const int lane = tid & 31;
    const int n  = lane & 15;      // B/C column within tile; also A row for loads
    const int hh = lane >> 4;      // half-wave selector
    const int col = nbase + n;
    const float bias = bvec[col];
    const int tilesM = (batch + 15) >> 4;

    float den = 0.0f;
    float num[10];
#pragma unroll
    for (int j = 0; j < 10; ++j) num[j] = 0.0f;

    for (int mt = blockIdx.y * 8 + wave; mt < tilesM; mt += gridDim.y * 8) {
        const u16* arow = Xb + (size_t)idx[mt * 16 + n] * DPAD;
        __builtin_prefetch(arow, 0, 0);
        v8f acc = {};
#pragma unroll 5
        for (int k0 = 0; k0 < DPAD; k0 += 32) {
            Frag A, B;
            // 16-bit A 16x32 layout: lane holds K = [hh*8, +8) and [hh*8+16, +8)
            A.u[0] = *(const uint4*)(arow + k0 + hh * 8);
            A.u[1] = *(const uint4*)(arow + k0 + hh * 8 + 16);
            // 16-bit B 32x16 layout: lane (col n) holds K = [hh*8, +8) and [hh*8+16, +8)
            const u16* bp = &Bl[n * DPAD + k0 + hh * 8];
            B.u[0] = *(const uint4*)(bp);
            B.u[1] = *(const uint4*)(bp + 16);
            acc = __builtin_amdgcn_wmma_f32_16x16x32_bf16(
                false, A.v, false, B.v, (short)0, acc, false, false);
        }
        // Epilogue: C layout -> VGPR r is row hh*8+r, lane&15 is column.
#pragma unroll
        for (int r = 0; r < 8; ++r) {
            int rowid = mt * 16 + hh * 8 + r;
            float hvv = sigmoidf(acc[r] + bias);
            if (rowid < batch) {
                den += hvv * hvv;
                const float* ekr = ek + (size_t)rowid * 10;
#pragma unroll
                for (int j = 0; j < 10; ++j) num[j] += ekr[j] * hvv;
            }
        }
    }

    // combine the two half-waves (rows 0-7 with rows 8-15), then one atomic set
    den += __shfl_xor(den, 16, 32);
#pragma unroll
    for (int j = 0; j < 10; ++j) num[j] += __shfl_xor(num[j], 16, 32);
    if (hh == 0) {
        atomicAdd(&vden[col], den);
#pragma unroll
        for (int j = 0; j < 10; ++j) atomicAdd(&vnum[j * NCPAD + col], num[j]);
    }
}

// ---------- score + argmax + emit chosen column into d_out ----------
__global__ __launch_bounds__(128) void score_select(const float* __restrict__ vnum,
                                                    const float* __restrict__ vden,
                                                    const float* __restrict__ Wf,
                                                    const float* __restrict__ bvec,
                                                    float* __restrict__ dout,
                                                    int* __restrict__ best_ws, int k) {
    __shared__ float sc[NCAND];
    __shared__ int bshare;
    int t = threadIdx.x;
    for (int c = t; c < NCAND; c += 128) {
        float s = 0.0f;
        for (int j = 0; j < 10; ++j) { float nm = vnum[j * NCPAD + c]; s += nm * nm; }
        sc[c] = s * __builtin_amdgcn_rcpf(vden[c] * 10.0f);
    }
    __syncthreads();
    if (t == 0) {
        int best = 0; float bv = sc[0];
        for (int c = 1; c < NCAND; ++c) if (sc[c] > bv) { bv = sc[c]; best = c; }
        bshare = best; best_ws[0] = best;
    }
    __syncthreads();
    int best = bshare;
    for (int i = t; i < DIMR; i += 128) dout[i * MAXK + k] = Wf[(size_t)best * DIMR + i];
    if (t == 0) dout[DIMR * MAXK + k] = bvec[best];
}

// ---------- h_c = sigmoid(X @ W_c + b_c) over all 30000 rows ----------
__global__ __launch_bounds__(256) void hc_kernel(const u16* __restrict__ Xb,
                                                 const u16* __restrict__ Wt,
                                                 const float* __restrict__ bvec,
                                                 const int* __restrict__ best_ws,
                                                 float* __restrict__ H, int k) {
    int i = blockIdx.x * 256 + threadIdx.x;
    if (i >= N_ROWS) return;
    int best = best_ws[0];
    const u16* xr = Xb + (size_t)i * DPAD;
    const u16* wr = Wt + (size_t)best * DPAD;
    float acc = 0.0f;
    for (int kk = 0; kk < DPAD; kk += 8) {
        uint4 a = *(const uint4*)(xr + kk);
        uint4 b = *(const uint4*)(wr + kk);
        acc += bfl(a.x) * bfl(b.x) + bfh(a.x) * bfh(b.x);
        acc += bfl(a.y) * bfl(b.y) + bfh(a.y) * bfh(b.y);
        acc += bfl(a.z) * bfl(b.z) + bfh(a.z) * bfh(b.z);
        acc += bfl(a.w) * bfl(b.w) + bfh(a.w) * bfh(b.w);
    }
    H[(size_t)k * N_ROWS + i] = sigmoidf(acc + bvec[best]);
}

// ---------- Gram-Schmidt: r1[j] = Q[:,j] . h_c  (one block per j) ----------
__global__ __launch_bounds__(256) void proj_r1(const float* __restrict__ Q,
                                               const float* __restrict__ H,
                                               float* __restrict__ r1,
                                               float* __restrict__ Rm, int k) {
    int j = blockIdx.x;
    __shared__ float red[256];
    const float* q = Q + (size_t)j * N_ROWS;
    const float* h = H + (size_t)k * N_ROWS;
    float s = 0.0f;
    for (int i = threadIdx.x; i < N_ROWS; i += 256) s += q[i] * h[i];
    red[threadIdx.x] = s;
    __syncthreads();
    for (int o = 128; o > 0; o >>= 1) {
        if (threadIdx.x < o) red[threadIdx.x] += red[threadIdx.x + o];
        __syncthreads();
    }
    if (threadIdx.x == 0) { r1[j] = red[0]; Rm[j * MAXK + k] = red[0]; }
}

// ---------- t = h_c - Q r1, and ||t||^2 ----------
__global__ __launch_bounds__(256) void compute_t(const float* __restrict__ Q,
                                                 const float* __restrict__ H,
                                                 const float* __restrict__ r1,
                                                 float* __restrict__ tvec,
                                                 float* __restrict__ scal, int k) {
    __shared__ float red[256];
    int i = blockIdx.x * 256 + threadIdx.x;
    float t = 0.0f;
    if (i < N_ROWS) {
        t = H[(size_t)k * N_ROWS + i];
        for (int j = 0; j < k; ++j) t -= Q[(size_t)j * N_ROWS + i] * r1[j];
        tvec[i] = t;
    }
    red[threadIdx.x] = t * t;
    __syncthreads();
    for (int o = 128; o > 0; o >>= 1) {
        if (threadIdx.x < o) red[threadIdx.x] += red[threadIdx.x + o];
        __syncthreads();
    }
    if (threadIdx.x == 0) atomicAdd(&scal[0], red[0]);
}

__global__ void finalize_r2(float* scal, float* Rm, int k) {
    if (threadIdx.x == 0) {
        float r2 = sqrtf(scal[0]);
        scal[1] = r2;
        scal[2] = 1.0f / r2;
        Rm[k * MAXK + k] = r2;
    }
}

// ---------- q = t/r2, QT[k,:] = q . Y ----------
__global__ __launch_bounds__(256) void qnorm_qty(const float* __restrict__ tvec,
                                                 const float* __restrict__ Y,
                                                 const float* __restrict__ scal,
                                                 float* __restrict__ Q,
                                                 float* __restrict__ QT, int k) {
    __shared__ float s[10];
    int tid = threadIdx.x;
    if (tid < 10) s[tid] = 0.0f;
    __syncthreads();
    int i = blockIdx.x * 256 + tid;
    if (i < N_ROWS) {
        float q = tvec[i] * scal[2];
        Q[(size_t)k * N_ROWS + i] = q;
        for (int j = 0; j < 10; ++j) atomicAdd(&s[j], q * Y[(size_t)i * 10 + j]);
    }
    __syncthreads();
    if (tid < 10) atomicAdd(&QT[k * 10 + tid], s[tid]);
}

// ---------- beta = solve_triangular(R, QT); also write into d_out ----------
__global__ void solve_beta(const float* __restrict__ Rm, const float* __restrict__ QT,
                           float* __restrict__ beta, float* __restrict__ dbeta, int k) {
    int j = threadIdx.x;
    if (j >= 10) return;
    float x[MAXK];
    for (int i = k; i >= 0; --i) {
        float v = QT[i * 10 + j];
        for (int m = i + 1; m <= k; ++m) v -= Rm[i * MAXK + m] * x[m];
        x[i] = v / Rm[i * MAXK + i];
    }
    for (int i = 0; i <= k; ++i) {
        beta[i * 10 + j]  = x[i];
        dbeta[i * 10 + j] = x[i];
    }
}

// ---------------------------------------------------------------------------
extern "C" void kernel_launch(void* const* d_in, const int* in_sizes, int n_in,
                              void* d_out, int out_size, void* d_ws, size_t ws_size,
                              hipStream_t stream) {
    (void)in_sizes; (void)n_in; (void)out_size; (void)ws_size;
    const float* X = (const float*)d_in[0];
    const float* Y = (const float*)d_in[1];
    float* out = (float*)d_out;

    char* ws = (char*)d_ws;
    size_t off = 0;
    auto take = [&](size_t bytes) -> char* {
        char* p = ws + off;
        off += (bytes + 255) & ~(size_t)255;
        return p;
    };

    u16*   Xb   = (u16*)take((size_t)N_ROWS * DPAD * 2);   // 48 MB, L2-resident
    u16*   Wt   = (u16*)take((size_t)NCPAD * DPAD * 2);
    float* Wf   = (float*)take((size_t)NCAND * DIMR * 4);
    float* bvec = (float*)take(NCPAD * 4);
    int*   idx  = (int*)take(MAXPAD * 4);
    float* ek   = (float*)take((size_t)MAXPAD * 10 * 4);
    float* vnum = (float*)take(10 * NCPAD * 4);
    float* vden = (float*)take(NCPAD * 4);
    int*   best = (int*)take(16);
    float* scal = (float*)take(16);
    float* H    = (float*)take((size_t)MAXK * N_ROWS * 4);
    float* Q    = (float*)take((size_t)MAXK * N_ROWS * 4);
    float* tvec = (float*)take((size_t)N_ROWS * 4);
    float* r1   = (float*)take(MAXK * 4);
    float* Rm   = (float*)take(MAXK * MAXK * 4);
    float* QT   = (float*)take(MAXK * 10 * 4);
    float* beta = (float*)take(MAXK * 10 * 4);

    convert_x<<<N_ROWS, 256, 0, stream>>>(X, Xb);

    int batch = 4000;
    for (int k = 0; k < MAXK; ++k) {
        gen_wb<<<(NCPAD * DPAD + 255) / 256, 256, 0, stream>>>(Wt, Wf, bvec, k);
        gen_idx<<<(MAXPAD + 255) / 256, 256, 0, stream>>>(idx, k);
        ek_kernel<<<(batch + 255) / 256, 256, 0, stream>>>(Y, H, beta, idx, ek, k, batch);
        zero_acc<<<5, 256, 0, stream>>>(vnum, vden, scal, QT, k);
        dim3 g(NCPAD / 16, 32);
        cand_gemm<<<g, 256, 0, stream>>>(Xb, Wt, bvec, idx, ek, vnum, vden, batch);
        score_select<<<1, 128, 0, stream>>>(vnum, vden, Wf, bvec, out, best, k);
        hc_kernel<<<(N_ROWS + 255) / 256, 256, 0, stream>>>(Xb, Wt, bvec, best, H, k);
        if (k > 0) proj_r1<<<k, 256, 0, stream>>>(Q, H, r1, Rm, k);
        compute_t<<<(N_ROWS + 255) / 256, 256, 0, stream>>>(Q, H, r1, tvec, scal, k);
        finalize_r2<<<1, 32, 0, stream>>>(scal, Rm, k);
        qnorm_qty<<<(N_ROWS + 255) / 256, 256, 0, stream>>>(tvec, Y, scal, Q, QT, k);
        solve_beta<<<1, 32, 0, stream>>>(Rm, QT, beta, out + DIMR * MAXK + MAXK, k);
        batch += 2166;
    }
}
